// Qwen3Attention_2216203124795
// MI455X (gfx1250) — matrix operand order repaired and checked by reading, not compiled, not run on hardware
//
#include <hip/hip_runtime.h>
#include <hip/hip_bf16.h>

typedef __bf16 bf16_t;
typedef __attribute__((ext_vector_type(16))) __bf16 v16bf;
typedef __attribute__((ext_vector_type(8)))  __bf16 bf16x8;
typedef __attribute__((ext_vector_type(4)))  __bf16 bf16x4;
typedef __attribute__((ext_vector_type(8)))  float  v8f;
typedef __attribute__((ext_vector_type(4)))  float  f32x4;
typedef __attribute__((ext_vector_type(4)))  int    v4i;

#define HIDDEN 4096
#define NQ 32
#define NKV 8
#define HD 128
#define TOK 2048
#define QKV_N ((NQ + 2 * NKV) * HD) /* 6144 */
#define KT 64

// ---- CDNA5 async global->LDS path (guarded; falls back to sync staging) ----
#if defined(__gfx1250__) && __has_builtin(__builtin_amdgcn_global_load_async_to_lds_b128) && __has_builtin(__builtin_amdgcn_s_wait_asynccnt)
#define USE_ASYNC_LDS 1
#else
#define USE_ASYNC_LDS 0
#endif

#if USE_ASYNC_LDS
typedef __attribute__((address_space(1))) v4i* gv4i_p;  // global int4*
typedef __attribute__((address_space(3))) v4i* sv4i_p;  // LDS int4*
__device__ __forceinline__ void async_copy16(void* lds, const void* g) {
  // GLOBAL_LOAD_ASYNC_TO_LDS_B128: per-lane 16B, tracked by ASYNCcnt
  __builtin_amdgcn_global_load_async_to_lds_b128((gv4i_p)g, (sv4i_p)lds, 0, 0);
}
__device__ __forceinline__ void wait_async0() {
  __builtin_amdgcn_s_wait_asynccnt(0);
}
#else
__device__ __forceinline__ void wait_async0() {}
#endif

__device__ __forceinline__ v16bf join16(bf16x8 lo, bf16x8 hi) {
  return __builtin_shufflevector(lo, hi, 0, 1, 2, 3, 4, 5, 6, 7,
                                 8, 9, 10, 11, 12, 13, 14, 15);
}

// ---------------------------------------------------------------------------
// f32 -> bf16 cast, 4 elements per lane (n must be %4)
// ---------------------------------------------------------------------------
__global__ void f32_to_bf16_kernel(const float* __restrict__ src,
                                   bf16_t* __restrict__ dst, int n4) {
  int i = blockIdx.x * blockDim.x + threadIdx.x;
  int stride = gridDim.x * blockDim.x;
  for (; i < n4; i += stride) {
    f32x4 v = ((const f32x4*)src)[i];
    bf16x4 o;
    o[0] = (bf16_t)v[0];
    o[1] = (bf16_t)v[1];
    o[2] = (bf16_t)v[2];
    o[3] = (bf16_t)v[3];
    ((bf16x4*)dst)[i] = o;
  }
}

// ---------------------------------------------------------------------------
// NT GEMM: C[M][N] f32 = A[M][K] bf16 * B[N][K]^T bf16   (M,N % 128; K % 32)
// 256 threads = 8 waves; 128x128 C tile; BK=32; per-wave 64x32 (4x2 WMMA).
// Double-buffered LDS; async global->LDS staging when available.
// ---------------------------------------------------------------------------
__global__ void __launch_bounds__(256)
gemm_nt_bf16f32(const bf16_t* __restrict__ A, const bf16_t* __restrict__ B,
                float* __restrict__ C, int M, int N, int K) {
  __shared__ bf16_t Alds[2][128][40];
  __shared__ bf16_t Blds[2][128][40];
  const int tid  = threadIdx.x;
  const int lane = tid & 31;
  const int wave = tid >> 5;
  const int wm   = wave >> 2;  // 0..1
  const int wn   = wave & 3;   // 0..3
  const int m0   = blockIdx.y * 128;
  const int n0   = blockIdx.x * 128;
  const int lrow = tid >> 1;
  const int lcol = (tid & 1) * 16;
  const int hi   = lane >> 4;
  const int l15  = lane & 15;

  v8f acc[4][2] = {};

  auto stage = [&](int buf, int k0) {
    const bf16_t* ap = A + (size_t)(m0 + lrow) * K + k0 + lcol;
    const bf16_t* bp = B + (size_t)(n0 + lrow) * K + k0 + lcol;
#if USE_ASYNC_LDS
    async_copy16(&Alds[buf][lrow][lcol],     ap);
    async_copy16(&Alds[buf][lrow][lcol + 8], ap + 8);
    async_copy16(&Blds[buf][lrow][lcol],     bp);
    async_copy16(&Blds[buf][lrow][lcol + 8], bp + 8);
#else
    bf16x8 a0 = *(const bf16x8*)(ap);
    bf16x8 a1 = *(const bf16x8*)(ap + 8);
    bf16x8 b0 = *(const bf16x8*)(bp);
    bf16x8 b1 = *(const bf16x8*)(bp + 8);
    *(bf16x8*)&Alds[buf][lrow][lcol]     = a0;
    *(bf16x8*)&Alds[buf][lrow][lcol + 8] = a1;
    *(bf16x8*)&Blds[buf][lrow][lcol]     = b0;
    *(bf16x8*)&Blds[buf][lrow][lcol + 8] = b1;
#endif
    __builtin_prefetch(ap + 64, 0, 1);  // two BK slabs ahead
    __builtin_prefetch(bp + 64, 0, 1);
  };

  stage(0, 0);
  wait_async0();
  __syncthreads();

  int cur = 0;
  for (int k0 = 0; k0 < K; k0 += 32) {
    if (k0 + 32 < K) stage(cur ^ 1, k0 + 32);  // prefetch next slab into other buffer

    v16bf bfrag[2];
#pragma unroll
    for (int j = 0; j < 2; ++j) {
      int n = wn * 32 + j * 16 + l15;
      int kb2 = hi * 16;  // lanes 0-15: K 0..15, lanes 16-31: K 16..31
      bfrag[j] = join16(*(const bf16x8*)&Blds[cur][n][kb2],
                        *(const bf16x8*)&Blds[cur][n][kb2 + 8]);
    }
#pragma unroll
    for (int i = 0; i < 4; ++i) {
      int m  = wm * 64 + i * 16 + l15;
      int kb = hi * 16;  // lanes 0-15: K 0..15, lanes 16-31: K 16..31
      v16bf afrag = join16(*(const bf16x8*)&Alds[cur][m][kb],
                           *(const bf16x8*)&Alds[cur][m][kb + 8]);
#pragma unroll
      for (int j = 0; j < 2; ++j)
        acc[i][j] = __builtin_amdgcn_wmma_f32_16x16x32_bf16(
            false, afrag, false, bfrag[j], (short)0, acc[i][j], false, false);
    }

    wait_async0();   // our async writes into cur^1 are done
    __syncthreads(); // everyone finished reading cur; cur^1 published
    cur ^= 1;
  }

#pragma unroll
  for (int i = 0; i < 4; ++i)
#pragma unroll
    for (int j = 0; j < 2; ++j) {
      int col  = n0 + wn * 32 + j * 16 + l15;
      int rowb = m0 + wm * 64 + i * 16 + hi * 8;
#pragma unroll
      for (int r = 0; r < 8; ++r)
        C[(size_t)(rowb + r) * N + col] = acc[i][j][r];
    }
}

// ---------------------------------------------------------------------------
// RMSNorm + RoPE + bf16 pack. Grid: (TOK, 12), block 128 (4 waves),
// one wave per (token, head). Heads 0..31 -> Q (pre-scaled by HD^-0.5),
// 32..39 -> K, 40..47 -> V passthrough.
// ---------------------------------------------------------------------------
__global__ void __launch_bounds__(128)
qkv_post_kernel(const float* __restrict__ qkv, const float* __restrict__ qw,
                const float* __restrict__ kw, const int* __restrict__ positions,
                bf16_t* __restrict__ qb, bf16_t* __restrict__ kb,
                bf16_t* __restrict__ vb) {
  const int t    = blockIdx.x;
  const int head = blockIdx.y * 4 + (threadIdx.x >> 5);
  const int lane = threadIdx.x & 31;
  const float pos = (float)positions[t];
  const float* row = qkv + (size_t)t * QKV_N;

  if (head >= NQ + NKV) {  // V: plain cast
    int h = head - NQ - NKV;
    const float* x = row + (NQ + NKV) * HD + h * HD;
    bf16_t* o = vb + ((size_t)h * TOK + t) * HD;
    o[lane]      = (bf16_t)x[lane];
    o[lane + 32] = (bf16_t)x[lane + 32];
    o[lane + 64] = (bf16_t)x[lane + 64];
    o[lane + 96] = (bf16_t)x[lane + 96];
    return;
  }

  const bool is_q = head < NQ;
  const int h = is_q ? head : head - NQ;
  const float* x = row + (is_q ? h * HD : NQ * HD + h * HD);
  const float* w = is_q ? qw : kw;

  float x0 = x[lane], x1 = x[lane + 32], x2 = x[lane + 64], x3 = x[lane + 96];
  float ss = x0 * x0 + x1 * x1 + x2 * x2 + x3 * x3;
  ss += __shfl_xor(ss, 1, 32);
  ss += __shfl_xor(ss, 2, 32);
  ss += __shfl_xor(ss, 4, 32);
  ss += __shfl_xor(ss, 8, 32);
  ss += __shfl_xor(ss, 16, 32);
  float inv = rsqrtf(ss * (1.0f / (float)HD) + 1e-6f);
  x0 *= inv * w[lane];
  x1 *= inv * w[lane + 32];
  x2 *= inv * w[lane + 64];
  x3 *= inv * w[lane + 96];

  // RoPE: pair (j, j+64); this lane owns j=lane and j=lane+32.
  const float LN_BASE = 9.210340371976184f;  // ln(10000)
  float s1, c1, s2, c2;
  __sincosf(pos * __expf(-(float)(2 * lane) * (LN_BASE / (float)HD)), &s1, &c1);
  __sincosf(pos * __expf(-(float)(2 * (lane + 32)) * (LN_BASE / (float)HD)), &s2, &c2);
  float y0 = x0 * c1 - x2 * s1;
  float y2 = x2 * c1 + x0 * s1;
  float y1 = x1 * c2 - x3 * s2;
  float y3 = x3 * c2 + x1 * s2;

  const float sc = is_q ? 0.08838834764831845f : 1.0f;  // HD^-0.5 folded into Q
  bf16_t* o = (is_q ? qb : kb) + ((size_t)h * TOK + t) * HD;
  o[lane]      = (bf16_t)(y0 * sc);
  o[lane + 32] = (bf16_t)(y1 * sc);
  o[lane + 64] = (bf16_t)(y2 * sc);
  o[lane + 96] = (bf16_t)(y3 * sc);
}

// ---------------------------------------------------------------------------
// Flash attention, causal, GQA. Grid: (TOK/64, NQ), block 128 = 4 waves.
// Each wave owns 16 query rows; key tiles of KT=64 staged in LDS.
// ---------------------------------------------------------------------------
__global__ void __launch_bounds__(128)
attn_kernel(const bf16_t* __restrict__ qb,    // [NQ][TOK][HD], pre-scaled
            const bf16_t* __restrict__ kbuf,  // [NKV][TOK][HD]
            const bf16_t* __restrict__ vbuf,  // [NKV][TOK][HD]
            bf16_t* __restrict__ ob) {        // [TOK][NQ*HD]
  __shared__ bf16_t Klds[KT][HD + 8];
  __shared__ bf16_t Vlds[KT][HD + 8];
  __shared__ bf16_t Plds[4][16][KT + 8];

  const int qh   = blockIdx.y;
  const int kvh  = qh >> 2;
  const int tid  = threadIdx.x;
  const int lane = tid & 31;
  const int wave = tid >> 5;
  const int hi   = lane >> 4;
  const int l15  = lane & 15;
  const int m_tile = blockIdx.x * 64 + wave * 16;

  // Q fragments (A-layout): 16 rows x HD=128 -> 4 frags of 16x32
  v16bf qfrag[4];
  {
    const bf16_t* qp = qb + ((size_t)qh * TOK + (m_tile + l15)) * HD;
#pragma unroll
    for (int kf = 0; kf < 4; ++kf) {
      int d0 = kf * 32 + hi * 8;
      qfrag[kf] = join16(*(const bf16x8*)(qp + d0), *(const bf16x8*)(qp + d0 + 16));
    }
  }

  v8f o[8] = {};
  float m_i[8], l_i[8];
#pragma unroll
  for (int r = 0; r < 8; ++r) { m_i[r] = -1e30f; l_i[r] = 0.0f; }

  const int njt = blockIdx.x + 1;  // every tile has >=1 unmasked col per row
  for (int j = 0; j < njt; ++j) {
    const int kt0 = j * KT;
    {  // stage K,V tile: 128 threads, 64 rows x 128 halves each
      int row = tid >> 1;
      int off = (tid & 1) * 64;
      const bf16_t* kp = kbuf + ((size_t)kvh * TOK + kt0 + row) * HD + off;
      const bf16_t* vp = vbuf + ((size_t)kvh * TOK + kt0 + row) * HD + off;
#if USE_ASYNC_LDS
#pragma unroll
      for (int i = 0; i < 8; ++i) {
        async_copy16(&Klds[row][off + i * 8], kp + i * 8);
        async_copy16(&Vlds[row][off + i * 8], vp + i * 8);
      }
      wait_async0();
#else
#pragma unroll
      for (int i = 0; i < 8; ++i) {
        *(bf16x8*)&Klds[row][off + i * 8] = *(const bf16x8*)(kp + i * 8);
        *(bf16x8*)&Vlds[row][off + i * 8] = *(const bf16x8*)(vp + i * 8);
      }
#endif
    }
    __syncthreads();

    // S = Q K^T (16 x 64), scale already folded into Q
    v8f s[4];
#pragma unroll
    for (int n = 0; n < 4; ++n) {
      v8f sa = {};
#pragma unroll
      for (int kf = 0; kf < 4; ++kf) {
        int kb2 = hi * 8 + kf * 32;
        v16bf kfrag = join16(*(const bf16x8*)&Klds[n * 16 + l15][kb2],
                             *(const bf16x8*)&Klds[n * 16 + l15][kb2 + 16]);
        sa = __builtin_amdgcn_wmma_f32_16x16x32_bf16(
            false, qfrag[kf], false, kfrag, (short)0, sa, false, false);
      }
      int col = kt0 + n * 16 + l15;
#pragma unroll
      for (int r = 0; r < 8; ++r) {
        int rowg = m_tile + hi * 8 + r;
        if (col > rowg) sa[r] = -1e30f;  // causal mask
      }
      s[n] = sa;
    }

    // online softmax (row reductions stay within each 16-lane half)
    float mnew[8], scl[8];
#pragma unroll
    for (int r = 0; r < 8; ++r) {
      float v = fmaxf(fmaxf(s[0][r], s[1][r]), fmaxf(s[2][r], s[3][r]));
      v = fmaxf(v, __shfl_xor(v, 1, 32));
      v = fmaxf(v, __shfl_xor(v, 2, 32));
      v = fmaxf(v, __shfl_xor(v, 4, 32));
      v = fmaxf(v, __shfl_xor(v, 8, 32));
      mnew[r] = fmaxf(m_i[r], v);
      scl[r]  = __expf(m_i[r] - mnew[r]);
      m_i[r]  = mnew[r];
    }
    float lad[8] = {};
#pragma unroll
    for (int n = 0; n < 4; ++n)
#pragma unroll
      for (int r = 0; r < 8; ++r) {
        float p = __expf(s[n][r] - mnew[r]);
        lad[r] += p;
        Plds[wave][hi * 8 + r][n * 16 + l15] = (bf16_t)p;  // C-layout -> LDS
      }
#pragma unroll
    for (int r = 0; r < 8; ++r) {
      float v = lad[r];
      v += __shfl_xor(v, 1, 32);
      v += __shfl_xor(v, 2, 32);
      v += __shfl_xor(v, 4, 32);
      v += __shfl_xor(v, 8, 32);
      l_i[r] = l_i[r] * scl[r] + v;
    }
#pragma unroll
    for (int n = 0; n < 8; ++n)
#pragma unroll
      for (int r = 0; r < 8; ++r) o[n][r] *= scl[r];

    // reload P in A-layout (wave-private LDS scratch, no barrier needed)
    v16bf pfrag[2];
#pragma unroll
    for (int kf = 0; kf < 2; ++kf) {
      int kbb = kf * 32 + hi * 16;
      pfrag[kf] = join16(*(const bf16x8*)&Plds[wave][l15][kbb],
                         *(const bf16x8*)&Plds[wave][l15][kbb + 8]);
    }
    // O += P @ V
#pragma unroll
    for (int n = 0; n < 8; ++n)
#pragma unroll
      for (int kf = 0; kf < 2; ++kf) {
        int kbase = kf * 32 + hi * 16;
        v16bf vfrag;
#pragma unroll
        for (int jj = 0; jj < 16; ++jj)
          vfrag[jj] = Vlds[kbase + jj][n * 16 + l15];
        o[n] = __builtin_amdgcn_wmma_f32_16x16x32_bf16(
            false, pfrag[kf], false, vfrag, (short)0, o[n], false, false);
      }
    __syncthreads();
  }

  // normalize + write bf16 [TOK][NQ*HD]
#pragma unroll
  for (int n = 0; n < 8; ++n)
#pragma unroll
    for (int r = 0; r < 8; ++r) {
      int rowg = m_tile + hi * 8 + r;
      ob[(size_t)rowg * (NQ * HD) + qh * HD + n * 16 + l15] =
          (bf16_t)(o[n][r] / l_i[r]);
    }
}

// ---------------------------------------------------------------------------
extern "C" void kernel_launch(void* const* d_in, const int* in_sizes, int n_in,
                              void* d_out, int out_size, void* d_ws,
                              size_t ws_size, hipStream_t stream) {
  (void)in_sizes; (void)n_in; (void)out_size; (void)ws_size;
  const float* hidden = (const float*)d_in[0];
  const float* w_qkv  = (const float*)d_in[1];
  const float* q_norm = (const float*)d_in[2];
  const float* k_norm = (const float*)d_in[3];
  const float* w_o    = (const float*)d_in[4];
  const int*   pos    = (const int*)d_in[5];
  float* out = (float*)d_out;

  char* ws = (char*)d_ws;
  size_t off = 0;
  auto alloc = [&](size_t bytes) -> void* {
    void* p = ws + off;
    off += (bytes + 255) & ~(size_t)255;
    return p;
  };
  bf16_t* hb    = (bf16_t*)alloc((size_t)TOK * HIDDEN * 2);
  bf16_t* wqkvb = (bf16_t*)alloc((size_t)QKV_N * HIDDEN * 2);
  bf16_t* wob   = (bf16_t*)alloc((size_t)HIDDEN * NQ * HD * 2);
  float*  qkvf  = (float*)alloc((size_t)TOK * QKV_N * 4);
  bf16_t* qbuf  = (bf16_t*)alloc((size_t)NQ * TOK * HD * 2);
  bf16_t* kbuf  = (bf16_t*)alloc((size_t)NKV * TOK * HD * 2);
  bf16_t* vbuf  = (bf16_t*)alloc((size_t)NKV * TOK * HD * 2);
  bf16_t* abuf  = (bf16_t*)alloc((size_t)TOK * NQ * HD * 2);

  f32_to_bf16_kernel<<<1024, 256, 0, stream>>>(hidden, hb, TOK * HIDDEN / 4);
  f32_to_bf16_kernel<<<1024, 256, 0, stream>>>(w_qkv, wqkvb, QKV_N * HIDDEN / 4);
  f32_to_bf16_kernel<<<1024, 256, 0, stream>>>(w_o, wob, HIDDEN * NQ * HD / 4);

  gemm_nt_bf16f32<<<dim3(QKV_N / 128, TOK / 128), 256, 0, stream>>>(
      hb, wqkvb, qkvf, TOK, QKV_N, HIDDEN);
  qkv_post_kernel<<<dim3(TOK, 12), 128, 0, stream>>>(
      qkvf, q_norm, k_norm, pos, qbuf, kbuf, vbuf);
  attn_kernel<<<dim3(TOK / 64, NQ), 128, 0, stream>>>(qbuf, kbuf, vbuf, abuf);
  gemm_nt_bf16f32<<<dim3(HIDDEN / 128, TOK / 128), 256, 0, stream>>>(
      abuf, wob, out, TOK, HIDDEN, NQ * HD);
}
